// EdgeGAT_43456479101295
// MI455X (gfx1250) — matrix operand order repaired and checked by reading, not compiled, not run on hardware
//
#include <hip/hip_runtime.h>

// ---------------------------------------------------------------------------
// EdgeGAT on gfx1250 (MI455X): 3x GATConv(edge_dim) + BN + ELU
// Dense GEMMs via v_wmma_f32_16x16x32_bf16; edge passes via L2-resident
// gather/scatter with float atomics; edge-attr attention term algebraically
// reduced to a 16xH matrix (removes the E x 128 edge projection entirely).
// ---------------------------------------------------------------------------

typedef __attribute__((ext_vector_type(16))) __bf16 v16bf;
typedef __attribute__((ext_vector_type(8)))  __bf16 v8bf;
typedef __attribute__((ext_vector_type(8)))  float  v8f;

#define LDS_STRIDE 136   // bf16 elems per row (128 + pad), 272B: 16B aligned

// --- order-preserving float<->uint encoding for atomic max ---
__device__ __forceinline__ unsigned encf(float f) {
    unsigned u = __float_as_uint(f);
    return (u & 0x80000000u) ? ~u : (u | 0x80000000u);
}
__device__ __forceinline__ float decf(unsigned u) {
    return (u & 0x80000000u) ? __uint_as_float(u & 0x7fffffffu)
                             : __uint_as_float(~u);
}
#define ENC_NEG_INF 0x007FFFFFu   // encf(-inf)

// ---------------------------------------------------------------------------
// GEMM: XP[nrows,128] = A[nrows,128] @ W[128,128]  (f32 in, f32 out, bf16 MACs)
// Block: 128 threads = 4 waves; block covers 64 rows; each wave a 16x128 strip.
// ---------------------------------------------------------------------------
__global__ __launch_bounds__(128)
void k_gemm128_wmma(const float* __restrict__ A, const float* __restrict__ W,
                    float* __restrict__ XP, int nrows) {
    __shared__ __bf16 Xs[64 * LDS_STRIDE];          // row-major rows
    __shared__ __bf16 Wt[128 * LDS_STRIDE];         // Wt[col][k] (transposed)
    const int tid  = threadIdx.x;
    const int lane = tid & 31;
    const int wv   = tid >> 5;                      // wave 0..3
    const int blockRow = blockIdx.x * 64;

    // stage W (coalesced global read, transposed into LDS as bf16)
    for (int i = tid; i < 128 * 128; i += 128) {
        int k = i >> 7, col = i & 127;
        Wt[col * LDS_STRIDE + k] = (__bf16)W[i];
    }
    // stage 64 input rows (clamped)
    for (int i = tid; i < 64 * 128; i += 128) {
        int r = i >> 7, c = i & 127;
        int gr = blockRow + r; if (gr >= nrows) gr = nrows - 1;
        Xs[r * LDS_STRIDE + c] = (__bf16)A[(long)gr * 128 + c];
    }
    __syncthreads();

    const int M  = lane & 15;      // row / col within tile
    const int kh = lane >> 4;      // K-half select per ISA layout
    v8f acc[8] = {};               // 8 N-tiles of 16x16 f32

    const __bf16* xrow = &Xs[(wv * 16 + M) * LDS_STRIDE];
    #pragma unroll
    for (int t = 0; t < 4; ++t) {                       // K steps of 32
        union { v16bf v; v8bf h[2]; } af;
        const int kb = t * 32 + kh * 8;                 // A: 16-bit 16x32 layout
        af.h[0] = *(const v8bf*)(xrow + kb);            // K group 0
        af.h[1] = *(const v8bf*)(xrow + kb + 16);       // K group 1
        #pragma unroll
        for (int n = 0; n < 8; ++n) {                   // 8 col tiles
            const __bf16* wcol = &Wt[(n * 16 + M) * LDS_STRIDE];
            union { v16bf v; v8bf h[2]; } bf;
            const int kbb = t * 32 + kh * 8;            // B: 32x16 layout
            bf.h[0] = *(const v8bf*)(wcol + kbb);
            bf.h[1] = *(const v8bf*)(wcol + kbb + 16);
            acc[n] = __builtin_amdgcn_wmma_f32_16x16x32_bf16(
                         false, af.v, false, bf.v, (short)0, acc[n],
                         false, false);
        }
    }

    // D layout: lanes 0-15 -> M=v, lanes 16-31 -> M=v+8; N = lane&15
    #pragma unroll
    for (int n = 0; n < 8; ++n) {
        #pragma unroll
        for (int v = 0; v < 8; ++v) {
            int row = blockRow + wv * 16 + kh * 8 + v;
            if (row < nrows) XP[(long)row * 128 + n * 16 + M] = acc[n][v];
        }
    }
}

// ---------------------------------------------------------------------------
// small helpers
// ---------------------------------------------------------------------------
__global__ void k_zero(float* p, int n) {
    int i = blockIdx.x * blockDim.x + threadIdx.x;
    if (i < n) p[i] = 0.f;
}

// edge_attr column sums (for self-loop mean fill)
__global__ void k_ea_sum(const float* __restrict__ ea, float* __restrict__ sum16,
                         long E_) {
    __shared__ float part[16];
    if (threadIdx.x < 16) part[threadIdx.x] = 0.f;
    __syncthreads();
    float loc[16];
    #pragma unroll
    for (int d = 0; d < 16; ++d) loc[d] = 0.f;
    long stride = (long)gridDim.x * blockDim.x;
    for (long e = (long)blockIdx.x * blockDim.x + threadIdx.x; e < E_; e += stride) {
        const float* r = &ea[e * 16];
        #pragma unroll
        for (int d = 0; d < 16; ++d) loc[d] += r[d];
    }
    #pragma unroll
    for (int d = 0; d < 16; ++d) atomicAdd(&part[d], loc[d]);
    __syncthreads();
    if (threadIdx.x < 16) atomicAdd(&sum16[threadIdx.x], part[threadIdx.x]);
}

// reduced edge-attention matrices wr[d][h] = sum_c We[d, h*C+c]*ae[h,c],
// ea_mean, and self-loop alpha_e constants.  One block of 128 threads.
__global__ void k_prep_small(const float* ea_sum,
                             const float* We1, const float* ae1,
                             const float* We2, const float* ae2,
                             const float* We3, const float* ae3,
                             float* ea_mean, float* wr1, float* wr2, float* wr3,
                             float* aself1, float* aself2, float* aself3,
                             float invE) {
    int t = threadIdx.x;
    if (t < 16) ea_mean[t] = ea_sum[t] * invE;
    if (t < 64) {                       // layer 1: [16][4]
        int d = t >> 2, h = t & 3; float acc = 0.f;
        for (int c = 0; c < 32; ++c) acc += We1[d * 128 + h * 32 + c] * ae1[h * 32 + c];
        wr1[t] = acc;
    } else if (t < 128) {               // layer 2: [16][4]
        int i = t - 64; int d = i >> 2, h = i & 3; float acc = 0.f;
        for (int c = 0; c < 32; ++c) acc += We2[d * 128 + h * 32 + c] * ae2[h * 32 + c];
        wr2[i] = acc;
    }
    if (t < 16) {                       // layer 3: [16][1]
        float acc = 0.f;
        for (int c = 0; c < 2; ++c) acc += We3[t * 2 + c] * ae3[c];
        wr3[t] = acc;
    }
    __syncthreads();
    if (t < 4)       { float a = 0.f; for (int d = 0; d < 16; ++d) a += ea_mean[d] * wr1[d * 4 + t];     aself1[t] = a; }
    else if (t < 8)  { int h = t - 4; float a = 0.f; for (int d = 0; d < 16; ++d) a += ea_mean[d] * wr2[d * 4 + h]; aself2[h] = a; }
    else if (t == 8) { float a = 0.f; for (int d = 0; d < 16; ++d) a += ea_mean[d] * wr3[d];             aself3[0] = a; }
}

// per-node: attention src/dst scores, init max/sum accumulators, bias-init out
__global__ void k_node_prep(const float* __restrict__ xp,
                            const float* __restrict__ a_s, const float* __restrict__ a_d,
                            const float* __restrict__ bias,
                            float* __restrict__ asrc, float* __restrict__ adst,
                            unsigned* __restrict__ M, float* __restrict__ S,
                            float* __restrict__ out, int Nn, int H_, int C_) {
    int n = blockIdx.x * blockDim.x + threadIdx.x;
    if (n >= Nn) return;
    const int HC_ = H_ * C_;
    for (int h = 0; h < H_; ++h) {
        float ss = 0.f, dd = 0.f;
        for (int c = 0; c < C_; ++c) {
            float v = xp[(long)n * HC_ + h * C_ + c];
            ss += v * a_s[h * C_ + c];
            dd += v * a_d[h * C_ + c];
        }
        asrc[(long)n * H_ + h] = ss;
        adst[(long)n * H_ + h] = dd;
        M[(long)n * H_ + h] = ENC_NEG_INF;
        S[(long)n * H_ + h] = 0.f;
    }
    for (int c = 0; c < HC_; ++c) out[(long)n * HC_ + c] = bias[c];
}

// pass 1: alpha = leaky(asrc[s]+adst[d]+eattr.wr), segment-max over dst
__global__ void k_edge_alpha_max(const int* __restrict__ src, const int* __restrict__ dst,
                                 const float* __restrict__ ea,
                                 const float* __restrict__ wr, const float* __restrict__ aself,
                                 const float* __restrict__ asrc, const float* __restrict__ adst,
                                 float* __restrict__ alpha, unsigned* __restrict__ M,
                                 long E_, int Nn, int H_) {
    long e = (long)blockIdx.x * blockDim.x + threadIdx.x;
    if (e >= E_ + Nn) return;
    int s, d;
    float aeh[4];
    if (e < E_) {
        s = src[e]; d = dst[e];
        const float* er = &ea[e * 16];
        for (int h = 0; h < H_; ++h) {
            float acc = 0.f;
            #pragma unroll
            for (int k = 0; k < 16; ++k) acc += er[k] * wr[k * H_ + h];
            aeh[h] = acc;
        }
    } else {
        s = d = (int)(e - E_);
        for (int h = 0; h < H_; ++h) aeh[h] = aself[h];
    }
    for (int h = 0; h < H_; ++h) {
        float a = asrc[(long)s * H_ + h] + adst[(long)d * H_ + h] + aeh[h];
        a = (a > 0.f) ? a : 0.2f * a;                       // leaky_relu
        alpha[e * H_ + h] = a;
        atomicMax(&M[(long)d * H_ + h], encf(a));
    }
}

// pass 2: ex = exp(alpha - max[dst]); segment-sum over dst
__global__ void k_edge_exp_sum(const int* __restrict__ dst,
                               float* __restrict__ alpha, const unsigned* __restrict__ M,
                               float* __restrict__ S, long E_, int Nn, int H_) {
    long e = (long)blockIdx.x * blockDim.x + threadIdx.x;
    if (e >= E_ + Nn) return;
    int d = (e < E_) ? dst[e] : (int)(e - E_);
    for (int h = 0; h < H_; ++h) {
        float m  = decf(M[(long)d * H_ + h]);
        float ex = __expf(alpha[e * H_ + h] - m);
        alpha[e * H_ + h] = ex;
        atomicAdd(&S[(long)d * H_ + h], ex);
    }
}

// pass 3 (HC=128): 32 lanes per edge, float4 gather + 4 atomic adds per lane.
// xp and out are L2-resident (25.6 MB each).
__global__ void k_edge_scatter128(const int* __restrict__ src, const int* __restrict__ dst,
                                  const float* __restrict__ xp, const float* __restrict__ alpha,
                                  const float* __restrict__ S, float* __restrict__ out,
                                  long E_, int Nn) {
    long idx = (long)blockIdx.x * blockDim.x + threadIdx.x;
    long e   = idx >> 5;
    int lane = (int)(idx & 31);
    if (e >= E_ + Nn) return;
    int s, d;
    if (e < E_) { s = src[e]; d = dst[e]; } else { s = d = (int)(e - E_); }
    int c4 = lane * 4;
    int h  = lane >> 3;                              // 32 ch per head
    float w = alpha[e * 4 + h] / (S[(long)d * 4 + h] + 1e-16f);
    const float4 v = *(const float4*)&xp[(long)s * 128 + c4];
    float* o = &out[(long)d * 128 + c4];
    atomicAdd(o + 0, v.x * w);
    atomicAdd(o + 1, v.y * w);
    atomicAdd(o + 2, v.z * w);
    atomicAdd(o + 3, v.w * w);
}

// pass 3 for layer 3 (H=1, 2 channels)
__global__ void k_edge_scatter2(const int* __restrict__ src, const int* __restrict__ dst,
                                const float* __restrict__ xp, const float* __restrict__ alpha,
                                const float* __restrict__ S, float* __restrict__ out,
                                long E_, int Nn) {
    long e = (long)blockIdx.x * blockDim.x + threadIdx.x;
    if (e >= E_ + Nn) return;
    int s, d;
    if (e < E_) { s = src[e]; d = dst[e]; } else { s = d = (int)(e - E_); }
    float w = alpha[e] / (S[d] + 1e-16f);
    atomicAdd(&out[(long)d * 2 + 0], xp[(long)s * 2 + 0] * w);
    atomicAdd(&out[(long)d * 2 + 1], xp[(long)s * 2 + 1] * w);
}

// BatchNorm(eval) + ELU (bias already folded into the accumulator init)
__global__ void k_bn_elu(const float* __restrict__ in,
                         const float* __restrict__ g, const float* __restrict__ be,
                         const float* __restrict__ rm, const float* __restrict__ rv,
                         float* __restrict__ out, long total) {
    long i = (long)blockIdx.x * blockDim.x + threadIdx.x;
    if (i >= total) return;
    int c = (int)(i & 127);
    float v = in[i];
    v = (v - rm[c]) * (g[c] * rsqrtf(rv[c] + 1e-5f)) + be[c];
    out[i] = (v > 0.f) ? v : (__expf(v) - 1.f);
}

// layer-3 projection: [N,128] @ [128,2]
__global__ void k_gemm_tiny(const float* __restrict__ hbuf, const float* __restrict__ W3,
                            float* __restrict__ xp3, int Nn) {
    int n = blockIdx.x * blockDim.x + threadIdx.x;
    if (n >= Nn) return;
    float a0 = 0.f, a1 = 0.f;
    for (int k = 0; k < 128; ++k) {
        float v = hbuf[(long)n * 128 + k];
        a0 += v * W3[k * 2 + 0];
        a1 += v * W3[k * 2 + 1];
    }
    xp3[(long)n * 2 + 0] = a0;
    xp3[(long)n * 2 + 1] = a1;
}

// ---------------------------------------------------------------------------
extern "C" void kernel_launch(void* const* d_in, const int* in_sizes, int n_in,
                              void* d_out, int out_size, void* d_ws, size_t ws_size,
                              hipStream_t stream) {
    const float* x    = (const float*)d_in[0];
    const int*   eidx = (const int*)  d_in[1];
    const float* ea   = (const float*)d_in[2];
    const float* W1   = (const float*)d_in[3];
    const float* as1  = (const float*)d_in[4];
    const float* ad1  = (const float*)d_in[5];
    const float* We1  = (const float*)d_in[6];
    const float* ae1  = (const float*)d_in[7];
    const float* b1   = (const float*)d_in[8];
    const float* g1   = (const float*)d_in[9];
    const float* be1  = (const float*)d_in[10];
    const float* rm1  = (const float*)d_in[11];
    const float* rv1  = (const float*)d_in[12];
    const float* W2   = (const float*)d_in[13];
    const float* as2  = (const float*)d_in[14];
    const float* ad2  = (const float*)d_in[15];
    const float* We2  = (const float*)d_in[16];
    const float* ae2  = (const float*)d_in[17];
    const float* b2   = (const float*)d_in[18];
    const float* g2   = (const float*)d_in[19];
    const float* be2  = (const float*)d_in[20];
    const float* rm2  = (const float*)d_in[21];
    const float* rv2  = (const float*)d_in[22];
    const float* W3   = (const float*)d_in[23];
    const float* as3  = (const float*)d_in[24];
    const float* ad3  = (const float*)d_in[25];
    const float* We3  = (const float*)d_in[26];
    const float* ae3  = (const float*)d_in[27];
    const float* b3   = (const float*)d_in[28];

    const int  N = in_sizes[0] / 128;       // 50000
    const long E = in_sizes[1] / 2;         // 1600000
    const int* srcp = eidx;
    const int* dstp = eidx + E;

    // ---- workspace carve (floats) ----
    float* base = (float*)d_ws;
    float* BUFH  = base;                base += (size_t)N * 128;    // h (layer input)
    float* BUFXP = base;                base += (size_t)N * 128;    // xp
    float* BUFO  = base;                base += (size_t)N * 128;    // segment-sum out
    float* ALPHA = base;                base += (size_t)(E + N) * 4;
    float* ASRC  = base;                base += (size_t)N * 4;
    float* ADST  = base;                base += (size_t)N * 4;
    unsigned* MB = (unsigned*)base;     base += (size_t)N * 4;
    float* SB    = base;                base += (size_t)N * 4;
    float* SM    = base;                                            // small region
    float* ea_sum  = SM;        float* ea_mean = SM + 16;
    float* wr1     = SM + 32;   float* wr2     = SM + 96;
    float* wr3     = SM + 160;
    float* aself1  = SM + 176;  float* aself2  = SM + 180;  float* aself3 = SM + 184;

    const long ET = E + N;                       // total edges incl. self loops
    const int  TB = 256;
    const int  gE   = (int)((ET + TB - 1) / TB);
    const int  gE32 = (int)((ET * 32 + TB - 1) / TB);
    const int  gN   = (N + TB - 1) / TB;
    const int  gNC  = (int)(((long)N * 128 + TB - 1) / TB);
    const int  gG   = (N + 63) / 64;

    // ---- precompute: edge-attr mean + reduced attention matrices ----
    k_zero<<<1, 32, 0, stream>>>(ea_sum, 16);
    k_ea_sum<<<1024, TB, 0, stream>>>(ea, ea_sum, E);
    k_prep_small<<<1, 128, 0, stream>>>(ea_sum, We1, ae1, We2, ae2, We3, ae3,
                                        ea_mean, wr1, wr2, wr3,
                                        aself1, aself2, aself3, 1.0f / (float)E);

    // ---- layer 1 ----
    k_gemm128_wmma<<<gG, 128, 0, stream>>>(x, W1, BUFXP, N);
    k_node_prep<<<gN, TB, 0, stream>>>(BUFXP, as1, ad1, b1, ASRC, ADST, MB, SB, BUFO, N, 4, 32);
    k_edge_alpha_max<<<gE, TB, 0, stream>>>(srcp, dstp, ea, wr1, aself1, ASRC, ADST, ALPHA, MB, E, N, 4);
    k_edge_exp_sum<<<gE, TB, 0, stream>>>(dstp, ALPHA, MB, SB, E, N, 4);
    k_edge_scatter128<<<gE32, TB, 0, stream>>>(srcp, dstp, BUFXP, ALPHA, SB, BUFO, E, N);
    k_bn_elu<<<gNC, TB, 0, stream>>>(BUFO, g1, be1, rm1, rv1, BUFH, (long)N * 128);

    // ---- layer 2 ----
    k_gemm128_wmma<<<gG, 128, 0, stream>>>(BUFH, W2, BUFXP, N);
    k_node_prep<<<gN, TB, 0, stream>>>(BUFXP, as2, ad2, b2, ASRC, ADST, MB, SB, BUFO, N, 4, 32);
    k_edge_alpha_max<<<gE, TB, 0, stream>>>(srcp, dstp, ea, wr2, aself2, ASRC, ADST, ALPHA, MB, E, N, 4);
    k_edge_exp_sum<<<gE, TB, 0, stream>>>(dstp, ALPHA, MB, SB, E, N, 4);
    k_edge_scatter128<<<gE32, TB, 0, stream>>>(srcp, dstp, BUFXP, ALPHA, SB, BUFO, E, N);
    k_bn_elu<<<gNC, TB, 0, stream>>>(BUFO, g2, be2, rm2, rv2, BUFH, (long)N * 128);

    // ---- layer 3 (H=1, C=2) -> d_out ----
    float* outp = (float*)d_out;
    k_gemm_tiny<<<gN, TB, 0, stream>>>(BUFH, W3, BUFXP, N);   // reuse BUFXP as [N,2]
    k_node_prep<<<gN, TB, 0, stream>>>(BUFXP, as3, ad3, b3, ASRC, ADST, MB, SB, outp, N, 1, 2);
    k_edge_alpha_max<<<gE, TB, 0, stream>>>(srcp, dstp, ea, wr3, aself3, ASRC, ADST, ALPHA, MB, E, N, 1);
    k_edge_exp_sum<<<gE, TB, 0, stream>>>(dstp, ALPHA, MB, SB, E, N, 1);
    k_edge_scatter2<<<gE, TB, 0, stream>>>(srcp, dstp, BUFXP, ALPHA, SB, outp, E, N);
}
